// TritonWindowAttention_19069654794897
// MI455X (gfx1250) — matrix operand order repaired and checked by reading, not compiled, not run on hardware
//
#include <hip/hip_runtime.h>

#define NTOK 64
#define DIM  512
#define NH   16
#define HD   32
#define NWIN 1024
#define TBL  225     // (2*8-1)^2
#define CPBH 512

typedef __attribute__((ext_vector_type(2)))  float  v2f;
typedef __attribute__((ext_vector_type(8)))  float  v8f;
typedef __attribute__((ext_vector_type(8)))  __bf16 v8bf;
typedef __attribute__((ext_vector_type(16))) __bf16 v16bf;

__device__ __forceinline__ unsigned short f2bf(float f) {
  unsigned u = __builtin_bit_cast(unsigned, f);
  u += 0x7FFFu + ((u >> 16) & 1u);           // round-to-nearest-even
  return (unsigned short)(u >> 16);
}

// Generic LDS pointer -> wave-relative LDS byte offset (low 32 bits, ISA 10.2).
__device__ __forceinline__ unsigned ldsoff(const void* p) {
  return (unsigned)(unsigned long long)p;
}

// CDNA5 async DMA: global -> LDS, 16B per lane, tracked by ASYNCcnt.
__device__ __forceinline__ void async_b128(unsigned lds_off, const void* g) {
  asm volatile("global_load_async_to_lds_b128 %0, %1, off"
               :: "v"(lds_off), "v"(g) : "memory");
}
template <int N>
__device__ __forceinline__ void wait_async() {
  asm volatile("s_wait_asynccnt %0" :: "n"(N) : "memory");
}

// A-fragment for V_WMMA_F32_16X16X32_BF16: lane holds K=[16*hi,16*hi+16)
// of its row (contiguous order, matching the B fragments).
__device__ __forceinline__ v16bf load_a16(const unsigned short* row, int hi) {
  v8bf lo = *(const v8bf*)(row + 16 * hi);
  v8bf hh = *(const v8bf*)(row + 16 * hi + 8);
  return __builtin_shufflevector(lo, hh, 0,1,2,3,4,5,6,7,8,9,10,11,12,13,14,15);
}

// ---------------- CPB MLP: table (225,2) -> relu(512) -> (225,16) -------------
__global__ void cpb_table_kernel(const float* __restrict__ w1, const float* __restrict__ b1,
                                 const float* __restrict__ w2, const float* __restrict__ coords,
                                 float* __restrict__ tbl) {
  int t = threadIdx.x;
  if (t >= TBL) return;
  float cx = coords[t * 2 + 0], cy = coords[t * 2 + 1];
  float acc[NH];
#pragma unroll
  for (int o = 0; o < NH; ++o) acc[o] = 0.f;
  for (int j = 0; j < CPBH; ++j) {
    float hv = fmaxf(cx * w1[j * 2] + cy * w1[j * 2 + 1] + b1[j], 0.f);
#pragma unroll
    for (int o = 0; o < NH; ++o) acc[o] += hv * w2[o * CPBH + j];
  }
#pragma unroll
  for (int o = 0; o < NH; ++o) tbl[t * NH + o] = acc[o];
}

// bias[h][i][j] = 16*sigmoid(tbl[idx[i*64+j]][h])
__global__ void cpb_gather_kernel(const float* __restrict__ tbl, const int* __restrict__ idx,
                                  float* __restrict__ biasTab) {
  int g = blockIdx.x * 256 + threadIdx.x;   // 16*4096 total
  int ij = g & 4095, h = g >> 12;
  float v = tbl[idx[ij] * NH + h];
  biasTab[h * 4096 + ij] = 16.f / (1.f + __expf(-v));
}

// ---------------- QKV GEMM: (65536x512)@(512x1536)^T + bias -> bf16 q/k/v ----
// Double-buffered LDS tiles filled with async global->LDS DMA.
__global__ __launch_bounds__(256) void qkv_gemm_kernel(
    const float* __restrict__ x, const float* __restrict__ w,
    const float* __restrict__ qb, const float* __restrict__ vb,
    unsigned short* __restrict__ qo, unsigned short* __restrict__ ko,
    unsigned short* __restrict__ vo) {
  __shared__ float xs[2][64 * 32];
  __shared__ float wt[2][64 * 32];
  const int tid = threadIdx.x;
  const int wv = tid >> 5, lane = tid & 31, lm = lane & 15, hi = lane >> 4;
  const int mt = wv & 3, ntb = wv >> 2;          // 8 waves: 4 M-tiles x 2 N-pairs
  const int mbase = blockIdx.x * 64, nbase = blockIdx.y * 64;

  const unsigned xoff0 = ldsoff(&xs[0][0]), xoff1 = ldsoff(&xs[1][0]);
  const unsigned woff0 = ldsoff(&wt[0][0]), woff1 = ldsoff(&wt[1][0]);

  auto issue = [&](int kc, int buf) {            // 4 async b128 per thread (8KB+8KB / WG)
    unsigned xo = buf ? xoff1 : xoff0;
    unsigned wo = buf ? woff1 : woff0;
#pragma unroll
    for (int i = 0; i < 2; ++i) {
      int v = tid * 2 + i;
      int row = v >> 3, k4 = v & 7;
      async_b128(xo + v * 16, x + (size_t)(mbase + row) * DIM + kc * 32 + k4 * 4);
      async_b128(wo + v * 16, w + (size_t)(nbase + row) * DIM + kc * 32 + k4 * 4);
    }
  };

  issue(0, 0);
  v8f c0 = {}; v8f c1 = {};
  for (int kc = 0; kc < DIM / 32; ++kc) {
    const int cur = kc & 1;
    if (kc + 1 < DIM / 32) {
      issue(kc + 1, cur ^ 1);                    // stream next chunk during compute
      { int v = tid * 2, row = v >> 3, k4 = v & 7;   // warm L2 two chunks ahead
        if (kc + 2 < DIM / 32)
          __builtin_prefetch(x + (size_t)(mbase + row) * DIM + (kc + 2) * 32 + k4 * 4, 0, 1);
      }
      wait_async<4>();                           // chunk kc landed (in-order done)
    } else {
      wait_async<0>();
    }
    __syncthreads();
    const float* xc = xs[cur];
    const float* wc = wt[cur];
#pragma unroll
    for (int ks = 0; ks < 8; ++ks) {
      int k = ks * 4 + 2 * hi;                   // f32 A 16x4 layout
      v2f a  = *(const v2f*)(xc + (mt * 16 + lm) * 32 + k);
      v2f b0 = *(const v2f*)(wc + (ntb * 32 + lm) * 32 + k);
      v2f b1 = *(const v2f*)(wc + (ntb * 32 + 16 + lm) * 32 + k);
      c0 = __builtin_amdgcn_wmma_f32_16x16x4_f32(false, a, false, b0, (short)0, c0, false, false);
      c1 = __builtin_amdgcn_wmma_f32_16x16x4_f32(false, a, false, b1, (short)0, c1, false, false);
    }
    __syncthreads();                             // all reads done before buffer reuse
  }
#pragma unroll
  for (int half = 0; half < 2; ++half) {
    v8f c = half ? c1 : c0;
    int gcol = nbase + ntb * 32 + half * 16 + lm;
    float bias = (gcol < DIM) ? qb[gcol] : ((gcol < 2 * DIM) ? 0.f : vb[gcol - 2 * DIM]);
    int which = gcol >> 9, h = (gcol >> 5) & 15, d = gcol & 31;
    unsigned short* dst = (which == 0) ? qo : ((which == 1) ? ko : vo);
#pragma unroll
    for (int j = 0; j < 8; ++j) {
      int token = mbase + mt * 16 + j + 8 * hi;
      int b = token >> 6, n = token & 63;
      dst[(size_t)((b * NH + h) * NTOK + n) * HD + d] = f2bf(c[j] + bias);
    }
  }
}

// ---------------- fused attention + output projection, 1 WG per window -------
__global__ __launch_bounds__(512) void attn_proj_kernel(
    const unsigned short* __restrict__ qg, const unsigned short* __restrict__ kg,
    const unsigned short* __restrict__ vg, const float* __restrict__ biasTab,
    const float* __restrict__ lsc, const float* __restrict__ pjw,
    const float* __restrict__ pjb, float* __restrict__ out) {
  extern __shared__ char smem[];
  float* os   = (float*)smem;                    // 64*512 f32 attn-out accumulator
  float* smat = os + 64 * 512;                   // 64*64 f32 scores
  float* pw   = smat + 64 * 64;                  // 2 x 64*32 f32 proj_w chunks
  unsigned short* qs  = (unsigned short*)(pw + 2 * 64 * 32);  // 64*32 bf16
  unsigned short* ks2 = qs + 64 * 32;                         // 64*32 bf16
  unsigned short* vt  = ks2 + 64 * 32;                        // 32*64 bf16 (V^T)
  unsigned short* ps  = vt + 64 * 32;                         // 64*64 bf16 probs

  const int tid = threadIdx.x;
  const int wv = tid >> 5, lane = tid & 31, lm = lane & 15, hi = lane >> 4;
  const int win = blockIdx.x;

  for (int h = 0; h < NH; ++h) {
    __syncthreads();
    const size_t base = ((size_t)win * NH + h) * (NTOK * HD);
    if (tid < 256) async_b128(ldsoff(qs) + tid * 16, (const char*)(qg + base) + tid * 16);
    else           async_b128(ldsoff(ks2) + (tid - 256) * 16, (const char*)(kg + base) + (tid - 256) * 16);
    const unsigned short* vsrc = vg + base;
#pragma unroll
    for (int i = 0; i < 4; ++i) {                // transpose V -> vt[d][token]
      int e = tid + i * 512;
      vt[(e & 31) * 64 + (e >> 5)] = vsrc[e];
    }
    wait_async<0>();
    __syncthreads();

    {  // S = Q K^T : 16 tiles of 16x16, one bf16 WMMA (K=32) per wave
      const int mt = wv >> 2, nt = wv & 3;
      v16bf a = load_a16(qs + (mt * 16 + lm) * HD, hi);
      v16bf b = *(const v16bf*)(ks2 + (nt * 16 + lm) * HD + 16 * hi);
      v8f c = {};
      c = __builtin_amdgcn_wmma_f32_16x16x32_bf16(false, a, false, b, (short)0, c, false, false);
      float ls = lsc[h];
      const float* bt = biasTab + h * 4096;
      int nn = nt * 16 + lm;
#pragma unroll
      for (int j = 0; j < 8; ++j) {
        int m = mt * 16 + j + 8 * hi;
        smat[m * 64 + nn] = c[j] * ls + bt[m * 64 + nn];
      }
    }
    __syncthreads();

    {  // row softmax: 8 lanes per row, width-8 shuffle reductions (wave32)
      int r = tid >> 3, sub = tid & 7;
      const float* rowp = smat + r * 64 + sub * 8;
      float v[8], mx = -3.0e38f;
#pragma unroll
      for (int i = 0; i < 8; ++i) { v[i] = rowp[i]; mx = fmaxf(mx, v[i]); }
      for (int m = 1; m < 8; m <<= 1) mx = fmaxf(mx, __shfl_xor(mx, m, 8));
      float s = 0.f;
#pragma unroll
      for (int i = 0; i < 8; ++i) { v[i] = __expf(v[i] - mx); s += v[i]; }
      for (int m = 1; m < 8; m <<= 1) s += __shfl_xor(s, m, 8);
      float inv = 1.f / s;
      unsigned short* prow = ps + r * 64 + sub * 8;
#pragma unroll
      for (int i = 0; i < 8; ++i) prow[i] = f2bf(v[i] * inv);
    }
    __syncthreads();

    if (wv < 8) {  // O = P V : 8 tiles (4M x 2N), K=64 -> 2 chained bf16 WMMAs
      const int mt = wv >> 1, nt = wv & 1;
      v8f c = {};
#pragma unroll
      for (int kc = 0; kc < 2; ++kc) {
        v16bf a = load_a16(ps + (mt * 16 + lm) * 64 + kc * 32, hi);
        v16bf b = *(const v16bf*)(vt + (nt * 16 + lm) * 64 + kc * 32 + 16 * hi);
        c = __builtin_amdgcn_wmma_f32_16x16x32_bf16(false, a, false, b, (short)0, c, false, false);
      }
      int col = h * HD + nt * 16 + lm;
#pragma unroll
      for (int j = 0; j < 8; ++j) {
        int m = mt * 16 + j + 8 * hi;
        os[m * DIM + col] = c[j];
      }
    }
  }
  __syncthreads();

  // projection: out(64x512) = os(64x512) @ pjw^T + pjb, f32 WMMA.
  // proj_w streamed with double-buffered async global->LDS DMA (128 chunks).
  const int mt = wv >> 2, nt = wv & 3;
  const unsigned pwoff0 = ldsoff(pw), pwoff1 = ldsoff(pw) + 64 * 32 * 4;
  auto issue_pw = [&](int chunk) {               // 1 async b128 per thread (8KB / WG)
    int nblk = chunk >> 4, kc = chunk & 15;
    int row = tid >> 3, k4 = tid & 7;
    async_b128(((chunk & 1) ? pwoff1 : pwoff0) + tid * 16,
               pjw + (size_t)(nblk * 64 + row) * DIM + kc * 32 + k4 * 4);
  };
  issue_pw(0);
  v8f c = {};
  for (int chunk = 0; chunk < 128; ++chunk) {
    const int kc = chunk & 15, nblk = chunk >> 4;
    if (chunk + 1 < 128) { issue_pw(chunk + 1); wait_async<1>(); }
    else                 { wait_async<0>(); }
    __syncthreads();
    const float* pwc = (chunk & 1) ? (pw + 64 * 32) : pw;
    if (kc == 0) { v8f z = {}; c = z; }
#pragma unroll
    for (int ks = 0; ks < 8; ++ks) {
      int k = ks * 4 + 2 * hi;
      v2f a = *(const v2f*)(os + (mt * 16 + lm) * DIM + kc * 32 + k);
      v2f b = *(const v2f*)(pwc + (nt * 16 + lm) * 32 + k);
      c = __builtin_amdgcn_wmma_f32_16x16x4_f32(false, a, false, b, (short)0, c, false, false);
    }
    if (kc == 15) {
      int col = nblk * 64 + nt * 16 + lm;
      float bias = pjb[col];
#pragma unroll
      for (int j = 0; j < 8; ++j) {
        int tok = win * 64 + mt * 16 + j + 8 * hi;
        out[(size_t)tok * DIM + col] = c[j] + bias;
      }
    }
    __syncthreads();
  }
}

extern "C" void kernel_launch(void* const* d_in, const int* in_sizes, int n_in,
                              void* d_out, int out_size, void* d_ws, size_t ws_size,
                              hipStream_t stream) {
  const float* x    = (const float*)d_in[0];
  const float* qkvw = (const float*)d_in[1];
  const float* qb   = (const float*)d_in[2];
  const float* vb   = (const float*)d_in[3];
  const float* lsc  = (const float*)d_in[4];
  const float* w1   = (const float*)d_in[5];
  const float* b1   = (const float*)d_in[6];
  const float* w2   = (const float*)d_in[7];
  const float* pjw  = (const float*)d_in[8];
  const float* pjb  = (const float*)d_in[9];
  const float* rct  = (const float*)d_in[10];
  const int*   rpi  = (const int*)d_in[11];
  float* out = (float*)d_out;

  const size_t QSZ = (size_t)NWIN * NH * NTOK * HD;     // 33.5M elems per tensor
  unsigned short* qws = (unsigned short*)d_ws;
  unsigned short* kws = qws + QSZ;
  unsigned short* vws = kws + QSZ;
  float* tbl     = (float*)(vws + QSZ);
  float* biasTab = tbl + TBL * NH;

  cpb_table_kernel<<<1, 256, 0, stream>>>(w1, b1, w2, rct, tbl);
  cpb_gather_kernel<<<256, 256, 0, stream>>>(tbl, rpi, biasTab);
  qkv_gemm_kernel<<<dim3(NWIN * NTOK / 64, 3 * DIM / 64), 256, 0, stream>>>(
      x, qkvw, qb, vb, qws, kws, vws);
  size_t smembytes = (size_t)(64 * 512 + 64 * 64 + 2 * 64 * 32) * sizeof(float) +
                     (size_t)(64 * 32 * 3 + 64 * 64) * sizeof(unsigned short);  // 184320
  attn_proj_kernel<<<NWIN, 512, smembytes, stream>>>(
      qws, kws, vws, biasTab, lsc, pjw, pjb, out);
}
